// DecoderBlockMoE_18253611008462
// MI455X (gfx1250) — matrix operand order repaired and checked by reading, not compiled, not run on hardware
//
#include <hip/hip_runtime.h>
#include <hip/hip_bf16.h>

typedef __bf16 bf16;
typedef __attribute__((ext_vector_type(16))) __bf16 v16bf;
typedef __attribute__((ext_vector_type(8)))  float  v8f;
typedef int v4i __attribute__((ext_vector_type(4)));
typedef __attribute__((address_space(1))) v4i as1_v4i;
typedef __attribute__((address_space(3))) v4i as3_v4i;

#define Bc   2
#define Sc   2048
#define Dc   1024
#define Hc   16
#define HDc  64
#define Ec   8
#define FFc  2048
#define Tc   4096
#define CAPc 512
#define WINc 1024

// ---------------------------------------------------------------------------
// CDNA5 async global->LDS staging (ASYNCcnt) with sync fallback
// ---------------------------------------------------------------------------
#if __has_builtin(__builtin_amdgcn_global_load_async_to_lds_b128)
#define GEMM_ASYNC 1
#else
#define GEMM_ASYNC 0
#endif

#if GEMM_ASYNC
#if __has_builtin(__builtin_amdgcn_s_wait_asynccnt)
#define ASYNC_WAIT(n) __builtin_amdgcn_s_wait_asynccnt(n)
#else
#define ASYNC_WAIT(n) asm volatile("s_wait_asynccnt %0" ::"i"(n) : "memory")
#endif
#else
#define ASYNC_WAIT(n) do {} while (0)
#endif

__device__ __forceinline__ void stage16(const bf16* g, bf16* l) {
#if GEMM_ASYNC
  __builtin_amdgcn_global_load_async_to_lds_b128(
      (as1_v4i*)(void*)g, (as3_v4i*)(void*)l, 0, 0);
#else
  *reinterpret_cast<uint4*>(l) = *reinterpret_cast<const uint4*>(g);
#endif
}

// ---------------------------------------------------------------------------
// WMMA helpers (CDNA5 16x16x32 bf16, fp32 accumulate)
// ---------------------------------------------------------------------------
__device__ __forceinline__ v8f wmma_bf16(v16bf a, v16bf b, v8f c) {
  return __builtin_amdgcn_wmma_f32_16x16x32_bf16(false, a, false, b, (short)0, c, false, false);
}

// A fragment (16x32, 16-bit): lane 0-15 row m, K = kb..kb+7 and kb+16..kb+23
// with kb = k0 + 8*(lane>=16). Two contiguous 16B loads per lane.
__device__ __forceinline__ v16bf load_frag_a(const bf16* base, int ld, int row0, int k0) {
  int lane = threadIdx.x & 31;
  int m  = row0 + (lane & 15);
  int kb = k0 + ((lane >> 4) << 3);
  const bf16* p = base + (size_t)m * ld + kb;
  v16bf a;
  uint4* ap = reinterpret_cast<uint4*>(&a);
  ap[0] = *reinterpret_cast<const uint4*>(p);
  ap[1] = *reinterpret_cast<const uint4*>(p + 16);
  return a;
}

// B fragment (32x16, 16-bit) from transposed storage Bt[N][K]:
// lane holds column n = col0 + (lane&15), K = kb..kb+7 and kb+16..kb+23 with kb = k0 + 8*(lane>=16).
__device__ __forceinline__ v16bf load_frag_b(const bf16* base, int ld, int col0, int k0) {
  int lane = threadIdx.x & 31;
  int n  = col0 + (lane & 15);
  int kb = k0 + ((lane >> 4) << 3);
  const bf16* p = base + (size_t)n * ld + kb;
  v16bf b;
  uint4* bp = reinterpret_cast<uint4*>(&b);
  bp[0] = *reinterpret_cast<const uint4*>(p);
  bp[1] = *reinterpret_cast<const uint4*>(p + 16);
  return b;
}

// ---------------------------------------------------------------------------
// Batched bf16 WMMA GEMM: C[M,N](f32) = A[M,K](bf16) x Bt[N,K](bf16)
// Block = 8 waves, tile 256(M) x 64(N); wave tile 32x64 (8 WMMA / k-step).
// A/B k-slabs double-buffered in LDS via async global->LDS loads.
// Requires M % 256 == 0, N % 64 == 0, K % 32 == 0.
// ---------------------------------------------------------------------------
__global__ void __launch_bounds__(256)
k_gemm(const bf16* __restrict__ A, const bf16* __restrict__ Bt,
       float* __restrict__ C, const float* __restrict__ resid,
       const float* __restrict__ rowScale,
       int M, int N, int K, size_t sA, size_t sB, size_t sC, size_t sScale) {
  (void)M;
  int z = blockIdx.z;
  A  += (size_t)z * sA;
  Bt += (size_t)z * sB;
  C  += (size_t)z * sC;
  if (resid)    resid    += (size_t)z * sC;
  if (rowScale) rowScale += (size_t)z * sScale;

  int tid  = threadIdx.x;
  int wave = tid >> 5;
  int lane = tid & 31;
  int rowblk = blockIdx.y * 256;
  int col0   = blockIdx.x * 64;

  __shared__ bf16 lA[2][256 * 32];  // 32 KB
  __shared__ bf16 lB[2][64 * 32];   //  8 KB

  // Stage one 32-wide k-slab: 3 async (or sync) 16B chunks per thread,
  // uniform across all waves so a constant s_wait_asynccnt is correct.
  auto stage = [&](int buf, int k0) {
#pragma unroll
    for (int s = 0; s < 2; ++s) {           // A: 256x32 = 512 chunks
      int c = tid + s * 256;
      int r = c >> 1, kp = (c & 1) << 4;
      stage16(A + (size_t)(rowblk + r) * K + k0 + kp, &lA[buf][c * 16]);
    }
    {                                       // B: 64x32 = 128 chunks (dup x2)
      int c = tid & 127;
      int r = c >> 1, kp = (c & 1) << 4;
      stage16(Bt + (size_t)(col0 + r) * K + k0 + kp, &lB[buf][c * 16]);
    }
    __builtin_prefetch(A + (size_t)(rowblk + (tid >> 1)) * K + k0 + 32, 0, 1);
  };

  v8f z8 = {};
  v8f acc[2][4];
#pragma unroll
  for (int g = 0; g < 2; ++g)
#pragma unroll
    for (int t = 0; t < 4; ++t) acc[g][t] = z8;

  int nk = K >> 5;
  stage(0, 0);
  for (int kk = 0; kk < nk; ++kk) {
    int buf = kk & 1;
    bool more = (kk + 1) < nk;
    if (more) stage(buf ^ 1, (kk + 1) << 5);
    if (more) { ASYNC_WAIT(3); } else { ASYNC_WAIT(0); }
    __syncthreads();

    v16bf a0 = load_frag_a(&lA[buf][0], 32, wave * 32, 0);
    v16bf a1 = load_frag_a(&lA[buf][0], 32, wave * 32 + 16, 0);
#pragma unroll
    for (int t = 0; t < 4; ++t) {
      v16bf b = load_frag_b(&lB[buf][0], 32, t * 16, 0);
      acc[0][t] = wmma_bf16(a0, b, acc[0][t]);
      acc[1][t] = wmma_bf16(a1, b, acc[1][t]);
    }
    __syncthreads();
  }

  int hi8 = (lane >> 4) << 3;
  int l15 = lane & 15;
#pragma unroll
  for (int g = 0; g < 2; ++g) {
    int row0 = rowblk + wave * 32 + g * 16;
    float rs[8];
    if (rowScale) {
#pragma unroll
      for (int r = 0; r < 8; ++r) rs[r] = rowScale[row0 + r + hi8];
    }
#pragma unroll
    for (int t = 0; t < 4; ++t) {
#pragma unroll
      for (int r = 0; r < 8; ++r) {
        size_t rr = (size_t)(row0 + r + hi8);
        size_t cc = (size_t)(col0 + t * 16 + l15);
        float val = acc[g][t][r];
        if (rowScale) val *= rs[r];
        if (resid)    val += resid[rr * N + cc];
        C[rr * N + cc] = val;
      }
    }
  }
}

// ---------------------------------------------------------------------------
// Transpose + cast fp32 -> bf16 : dst[c*R + r] = src[r*C + c], batched over z
// ---------------------------------------------------------------------------
__global__ void k_transcast(const float* __restrict__ src, bf16* __restrict__ dst, int R, int Cc) {
  size_t base = (size_t)blockIdx.z * R * Cc;
  int idx = blockIdx.x * blockDim.x + threadIdx.x;
  if (idx >= R * Cc) return;
  int r = idx / Cc, c = idx % Cc;
  dst[base + (size_t)c * R + r] = (bf16)src[base + idx];
}

// ---------------------------------------------------------------------------
// RMSNorm: one row (D=1024) per block of 256 threads, bf16 output
// ---------------------------------------------------------------------------
__global__ void __launch_bounds__(256)
k_rmsnorm(const float* __restrict__ x, const float* __restrict__ w, bf16* __restrict__ out) {
  int row = blockIdx.x;
  const float* xr = x + (size_t)row * Dc;
  float ss = 0.f;
  for (int i = threadIdx.x; i < Dc; i += 256) { float v = xr[i]; ss += v * v; }
#pragma unroll
  for (int off = 16; off >= 1; off >>= 1) ss += __shfl_xor(ss, off, 32);
  __shared__ float red[8];
  if ((threadIdx.x & 31) == 0) red[threadIdx.x >> 5] = ss;
  __syncthreads();
  float tot = 0.f;
#pragma unroll
  for (int i = 0; i < 8; ++i) tot += red[i];
  float scale = rsqrtf(tot * (1.f / Dc) + 1e-6f);
  bf16* orow = out + (size_t)row * Dc;
  for (int i = threadIdx.x; i < Dc; i += 256) orow[i] = (bf16)(xr[i] * scale * w[i]);
}

// ---------------------------------------------------------------------------
// RoPE + layout: qkv[T,3D] f32 -> q_bf[bh,S,hd], k_bf[bh,S,hd], vT_bf[bh,hd,S]
// ---------------------------------------------------------------------------
__global__ void k_rope(const float* __restrict__ qkv, bf16* __restrict__ Qb,
                       bf16* __restrict__ Kb, bf16* __restrict__ Vt) {
  int gid = blockIdx.x * blockDim.x + threadIdx.x;   // T*D threads
  int t = gid >> 10;
  int d = gid & 1023;
  int b = t >> 11;
  int s = t & 2047;
  int h = d >> 6;
  int e = d & 63;
  const float* base = qkv + (size_t)t * (3 * Dc);
  float q = base[d];
  float k = base[Dc + d];
  float v = base[2 * Dc + d];
  int fi = e & 31;
  float inv = __powf(10000.f, -(float)fi * (1.f / 32.f));
  float ang = (float)s * inv;
  float cs = __cosf(ang), sn = __sinf(ang);
  int dp = (e < 32) ? d + 32 : d - 32;
  float qp = base[dp];
  float kp = base[Dc + dp];
  float qr = q * cs + ((e < 32) ? -qp : qp) * sn;
  float kr = k * cs + ((e < 32) ? -kp : kp) * sn;
  int bh = b * Hc + h;
  size_t qi = ((size_t)bh * Sc + s) * HDc + e;
  Qb[qi] = (bf16)qr;
  Kb[qi] = (bf16)kr;
  Vt[((size_t)bh * HDc + e) * Sc + s] = (bf16)v;
}

// ---------------------------------------------------------------------------
// Windowed flash attention: 1 wave per 16-query tile, 32-key steps.
// scores: 4 WMMAs/step, P·V: 4 WMMAs/step; P transposed C->A layout via LDS.
// ---------------------------------------------------------------------------
__global__ void __launch_bounds__(32)
k_attn(const bf16* __restrict__ Qb, const bf16* __restrict__ Kb,
       const bf16* __restrict__ Vt, bf16* __restrict__ O) {
  int bh = blockIdx.y;
  int b = bh >> 4, h = bh & 15;
  int i0 = blockIdx.x * 16;
  int lane = threadIdx.x;
  int hi8 = (lane >> 4) << 3;
  int l15 = lane & 15;

  const bf16* Q  = Qb + (size_t)bh * Sc * HDc;
  const bf16* Km = Kb + (size_t)bh * Sc * HDc;
  const bf16* Vm = Vt + (size_t)bh * HDc * Sc;

  v16bf aq0 = load_frag_a(Q, HDc, i0, 0);
  v16bf aq1 = load_frag_a(Q, HDc, i0, 32);

  v8f z8 = {};
  v8f ov[4];
#pragma unroll
  for (int t = 0; t < 4; ++t) ov[t] = z8;
  float mrow[8], lrow[8];
#pragma unroll
  for (int r = 0; r < 8; ++r) { mrow[r] = -1e30f; lrow[r] = 0.f; }

  __shared__ bf16 Pl[16][32];

  int jlo = i0 - (WINc - 1);
  if (jlo < 0) jlo = 0;
  jlo &= ~31;

  for (int j0 = jlo; j0 < i0 + 16; j0 += 32) {
    // ---- scores S[16x32] = Q x K^T ----
    v8f s0 = z8, s1 = z8;
    {
      v16bf bk = load_frag_b(Km, HDc, j0, 0);
      s0 = wmma_bf16(aq0, bk, s0);
      bk = load_frag_b(Km, HDc, j0 + 16, 0);
      s1 = wmma_bf16(aq0, bk, s1);
      bk = load_frag_b(Km, HDc, j0, 32);
      s0 = wmma_bf16(aq1, bk, s0);
      bk = load_frag_b(Km, HDc, j0 + 16, 32);
      s1 = wmma_bf16(aq1, bk, s1);
    }
    // ---- mask + online softmax ----
    float sv0[8], sv1[8];
    bool ok0[8], ok1[8];
#pragma unroll
    for (int r = 0; r < 8; ++r) {
      int irow = i0 + r + hi8;
      int jc0 = j0 + l15;
      int jc1 = jc0 + 16;
      ok0[r] = (jc0 <= irow) && (irow - jc0 < WINc);
      ok1[r] = (jc1 <= irow) && (irow - jc1 < WINc);
      sv0[r] = ok0[r] ? s0[r] * 0.125f : -1e30f;
      sv1[r] = ok1[r] ? s1[r] * 0.125f : -1e30f;
    }
#pragma unroll
    for (int r = 0; r < 8; ++r) {
      float rm = fmaxf(sv0[r], sv1[r]);
#pragma unroll
      for (int off = 1; off < 16; off <<= 1) rm = fmaxf(rm, __shfl_xor(rm, off, 32));
      float mn = fmaxf(mrow[r], rm);
      float alpha = __expf(mrow[r] - mn);
      float p0 = ok0[r] ? __expf(sv0[r] - mn) : 0.f;
      float p1 = ok1[r] ? __expf(sv1[r] - mn) : 0.f;
      float rsum = p0 + p1;
#pragma unroll
      for (int off = 1; off < 16; off <<= 1) rsum += __shfl_xor(rsum, off, 32);
      lrow[r] = lrow[r] * alpha + rsum;
      mrow[r] = mn;
      ov[0][r] *= alpha; ov[1][r] *= alpha; ov[2][r] *= alpha; ov[3][r] *= alpha;
      Pl[r + hi8][l15]      = (bf16)p0;
      Pl[r + hi8][l15 + 16] = (bf16)p1;
    }
    __syncthreads();
    // ---- O += P x V ----
    v16bf ap = load_frag_a(&Pl[0][0], 32, 0, 0);
#pragma unroll
    for (int t = 0; t < 4; ++t) {
      v16bf bv = load_frag_b(Vm, Sc, t * 16, j0);
      ov[t] = wmma_bf16(ap, bv, ov[t]);
    }
    __syncthreads();
  }

  // ---- epilogue: normalize, write bf16 [T, D] at column h*64 ----
#pragma unroll
  for (int r = 0; r < 8; ++r) {
    float inv = 1.f / lrow[r];
    int irow = i0 + r + hi8;
    size_t base = ((size_t)b * Sc + irow) * Dc + h * HDc + l15;
    O[base]      = (bf16)(ov[0][r] * inv);
    O[base + 16] = (bf16)(ov[1][r] * inv);
    O[base + 32] = (bf16)(ov[2][r] * inv);
    O[base + 48] = (bf16)(ov[3][r] * inv);
  }
}

// ---------------------------------------------------------------------------
// MoE gating: logits -> softmax(8) -> top-2 -> w_te, phi accumulation
// ---------------------------------------------------------------------------
__global__ void __launch_bounds__(128)
k_gate(const bf16* __restrict__ xn2, const bf16* __restrict__ gateT,
       float* __restrict__ wte, float* __restrict__ phi) {
  int t = blockIdx.x * blockDim.x + threadIdx.x;
  if (t >= Tc) return;
  const bf16* xr = xn2 + (size_t)t * Dc;
  float lg[Ec];
#pragma unroll
  for (int e = 0; e < Ec; ++e) {
    const bf16* g = gateT + (size_t)e * Dc;
    float acc = 0.f;
    for (int i = 0; i < Dc; ++i) acc += (float)xr[i] * (float)g[i];
    lg[e] = acc;
  }
  float mx = lg[0];
#pragma unroll
  for (int e = 1; e < Ec; ++e) mx = fmaxf(mx, lg[e]);
  float p[Ec], den = 0.f;
#pragma unroll
  for (int e = 0; e < Ec; ++e) { p[e] = __expf(lg[e] - mx); den += p[e]; }
  float rden = 1.f / den;
#pragma unroll
  for (int e = 0; e < Ec; ++e) p[e] *= rden;
  int i0 = 0;
#pragma unroll
  for (int e = 1; e < Ec; ++e) if (p[e] > p[i0]) i0 = e;
  int i1 = (i0 == 0) ? 1 : 0;
#pragma unroll
  for (int e = 0; e < Ec; ++e) if (e != i0 && p[e] > p[i1]) i1 = e;
#pragma unroll
  for (int e = 0; e < Ec; ++e) {
    wte[(size_t)t * Ec + e] = (e == i0 || e == i1) ? p[e] : 0.f;
    atomicAdd(&phi[e], p[e]);
  }
}

__global__ void k_zero(float* p, int n) {
  int i = blockIdx.x * blockDim.x + threadIdx.x;
  if (i < n) p[i] = 0.f;
}

__global__ void k_aux(const float* __restrict__ phi, float* __restrict__ out) {
  if (threadIdx.x == 0 && blockIdx.x == 0) {
    float s = 0.f;
#pragma unroll
    for (int e = 0; e < Ec; ++e) { float pm = phi[e] * (1.f / Tc); s += pm * pm; }
    out[(size_t)Tc * Dc] = (float)Ec * s;
  }
}

// exact per-expert top-512 via rank counting
__global__ void __launch_bounds__(256)
k_rank(const float* __restrict__ wte, int* __restrict__ capI, float* __restrict__ capV) {
  int e = blockIdx.y;
  int t = blockIdx.x * blockDim.x + threadIdx.x;
  float wv = wte[(size_t)t * Ec + e];
  int rank = 0;
  for (int u = 0; u < Tc; ++u) {
    float wu = wte[(size_t)u * Ec + e];
    rank += (wu > wv) || (wu == wv && u < t);
  }
  if (rank < CAPc) { capI[e * CAPc + rank] = t; capV[e * CAPc + rank] = wv; }
}

__global__ void k_gather(const bf16* __restrict__ xn2, const int* __restrict__ capI,
                         bf16* __restrict__ xe) {
  int gid = blockIdx.x * blockDim.x + threadIdx.x;  // E*CAP*(D/8)
  int dblk = gid & 127;
  int ec = gid >> 7;
  int tk = capI[ec];
  ((uint4*)xe)[(size_t)ec * 128 + dblk] = ((const uint4*)xn2)[(size_t)tk * 128 + dblk];
}

__global__ void k_swiglu(const float* __restrict__ hid, bf16* __restrict__ act) {
  int gid = blockIdx.x * blockDim.x + threadIdx.x;  // E*CAP*FF
  int f = gid & (FFc - 1);
  int ec = gid >> 11;
  const float* hb = hid + (size_t)ec * (2 * FFc);
  float h1 = hb[f], h2 = hb[FFc + f];
  float sig = 1.f / (1.f + __expf(-h2));
  act[(size_t)ec * FFc + f] = (bf16)(h1 * h2 * sig);
}

__global__ void k_scatter(const float* __restrict__ oe, const int* __restrict__ capI,
                          float* __restrict__ out) {
  int gid = blockIdx.x * blockDim.x + threadIdx.x;  // E*CAP*D
  int d = gid & (Dc - 1);
  int ec = gid >> 10;
  int tk = capI[ec];
  atomicAdd(&out[(size_t)tk * Dc + d], oe[gid]);
}

// ---------------------------------------------------------------------------
extern "C" void kernel_launch(void* const* d_in, const int* in_sizes, int n_in,
                              void* d_out, int out_size, void* d_ws, size_t ws_size,
                              hipStream_t stream) {
  (void)in_sizes; (void)n_in; (void)out_size; (void)ws_size;
  const float* x      = (const float*)d_in[0];
  const float* wq     = (const float*)d_in[1];
  const float* wk     = (const float*)d_in[2];
  const float* wv     = (const float*)d_in[3];
  const float* wo     = (const float*)d_in[4];
  const float* norm1  = (const float*)d_in[5];
  const float* norm2  = (const float*)d_in[6];
  const float* gate_w = (const float*)d_in[7];
  const float* w1     = (const float*)d_in[8];
  const float* w2     = (const float*)d_in[9];
  float* out = (float*)d_out;

  char* wsp = (char*)d_ws;
  auto alloc = [&](size_t bytes) -> char* {
    char* p = wsp;
    wsp += (bytes + 255) & ~(size_t)255;
    return p;
  };
  bf16*  wqkvT = (bf16*)alloc((size_t)3 * Dc * Dc * 2);
  bf16*  woT   = (bf16*)alloc((size_t)Dc * Dc * 2);
  bf16*  gateT = (bf16*)alloc((size_t)Ec * Dc * 2);
  bf16*  w1T   = (bf16*)alloc((size_t)Ec * 2 * FFc * Dc * 2);
  bf16*  w2T   = (bf16*)alloc((size_t)Ec * Dc * FFc * 2);
  bf16*  xn1   = (bf16*)alloc((size_t)Tc * Dc * 2);
  float* qkv   = (float*)alloc((size_t)Tc * 3 * Dc * 4);
  bf16*  qb    = (bf16*)alloc((size_t)Bc * Hc * Sc * HDc * 2 + 65536);
  bf16*  kb    = (bf16*)alloc((size_t)Bc * Hc * Sc * HDc * 2 + 65536);
  bf16*  vt    = (bf16*)alloc((size_t)Bc * Hc * HDc * Sc * 2 + 65536);
  bf16*  attnb = (bf16*)alloc((size_t)Tc * Dc * 2);
  bf16*  xn2   = (bf16*)alloc((size_t)Tc * Dc * 2);
  float* wte   = (float*)alloc((size_t)Tc * Ec * 4);
  float* phi   = (float*)alloc(256);
  int*   capI  = (int*)alloc((size_t)Ec * CAPc * 4);
  float* capV  = (float*)alloc((size_t)Ec * CAPc * 4);
  bf16*  xe    = (bf16*)alloc((size_t)Ec * CAPc * Dc * 2);
  float* hid   = (float*)alloc((size_t)Ec * CAPc * 2 * FFc * 4);
  bf16*  act   = (bf16*)alloc((size_t)Ec * CAPc * FFc * 2);
  float* oe    = (float*)alloc((size_t)Ec * CAPc * Dc * 4);

  // weight prep: transpose + cast to bf16
  k_transcast<<<dim3(4096, 1, 1), 256, 0, stream>>>(wq, wqkvT,               Dc, Dc);
  k_transcast<<<dim3(4096, 1, 1), 256, 0, stream>>>(wk, wqkvT + Dc * Dc,     Dc, Dc);
  k_transcast<<<dim3(4096, 1, 1), 256, 0, stream>>>(wv, wqkvT + 2 * Dc * Dc, Dc, Dc);
  k_transcast<<<dim3(4096, 1, 1), 256, 0, stream>>>(wo, woT, Dc, Dc);
  k_transcast<<<dim3(32, 1, 1),   256, 0, stream>>>(gate_w, gateT, Dc, Ec);
  k_transcast<<<dim3(16384, 1, Ec), 256, 0, stream>>>(w1, w1T, Dc, 2 * FFc);
  k_transcast<<<dim3(8192, 1, Ec),  256, 0, stream>>>(w2, w2T, FFc, Dc);

  // sublayer 1: rmsnorm -> fused QKV GEMM -> RoPE/layout -> flash attn -> out proj (+residual)
  k_rmsnorm<<<Tc, 256, 0, stream>>>(x, norm1, xn1);
  k_gemm<<<dim3(48, Tc / 256, 1), 256, 0, stream>>>(xn1, wqkvT, qkv, nullptr, nullptr,
                                                    Tc, 3 * Dc, Dc, 0, 0, 0, 0);
  k_rope<<<16384, 256, 0, stream>>>(qkv, qb, kb, vt);
  k_attn<<<dim3(Sc / 16, Bc * Hc, 1), 32, 0, stream>>>(qb, kb, vt, attnb);
  k_gemm<<<dim3(16, Tc / 256, 1), 256, 0, stream>>>(attnb, woT, out, x, nullptr,
                                                    Tc, Dc, Dc, 0, 0, 0, 0);

  // sublayer 2: rmsnorm -> gate/top-2 -> capacity select -> expert GEMMs -> scatter
  k_rmsnorm<<<Tc, 256, 0, stream>>>(out, norm2, xn2);
  k_zero<<<1, 32, 0, stream>>>(phi, Ec);
  k_gate<<<Tc / 128, 128, 0, stream>>>(xn2, gateT, wte, phi);
  k_aux<<<1, 32, 0, stream>>>(phi, out);
  k_rank<<<dim3(Tc / 256, Ec, 1), 256, 0, stream>>>(wte, capI, capV);
  k_gather<<<(Ec * CAPc * (Dc / 8)) / 256, 256, 0, stream>>>(xn2, capI, xe);
  k_gemm<<<dim3(2 * FFc / 64, CAPc / 256, Ec), 256, 0, stream>>>(
      xe, w1T, hid, nullptr, nullptr, CAPc, 2 * FFc, Dc,
      (size_t)CAPc * Dc, (size_t)2 * FFc * Dc, (size_t)CAPc * 2 * FFc, 0);
  k_swiglu<<<(Ec * CAPc * FFc) / 256, 256, 0, stream>>>(hid, act);
  k_gemm<<<dim3(Dc / 64, CAPc / 256, Ec), 256, 0, stream>>>(
      act, w2T, oe, nullptr, capV, CAPc, Dc, FFc,
      (size_t)CAPc * FFc, (size_t)Dc * FFc, (size_t)CAPc * Dc, (size_t)CAPc);
  k_scatter<<<(Ec * CAPc * Dc) / 256, 256, 0, stream>>>(oe, capI, out);
}
